// SinkhornOT_8916352106503
// MI455X (gfx1250) — matrix operand order repaired and checked by reading, not compiled, not run on hardware
//
#include <hip/hip_runtime.h>
#include <hip/hip_bf16.h>

typedef __attribute__((ext_vector_type(16))) _Float16 v16h;
typedef __attribute__((ext_vector_type(8)))  _Float16 v8h;
typedef __attribute__((ext_vector_type(8)))  float    v8f;

#define NM 4096
#define EPS_F 0.1f
#define LO_SCALE 2048.0f
#define LO_INV   (1.0f / 2048.0f)

// ---------------------------------------------------------------------------
// 1) Convert C (f32) -> Kf16 = exp(-C/eps) row-major, and KTf16 (transposed,
//    also row-major) via a 32x32 LDS tile. One-time 64MB read / 64MB write.
// ---------------------------------------------------------------------------
__global__ __launch_bounds__(256) void sink_convert(const float* __restrict__ C,
                                                    _Float16* __restrict__ K,
                                                    _Float16* __restrict__ KT) {
    __shared__ _Float16 tile[32][33];
    const int bx = blockIdx.x * 32, by = blockIdx.y * 32;
    const int tx = threadIdx.x, ty = threadIdx.y;  // 32 x 8
    #pragma unroll
    for (int r = 0; r < 32; r += 8) {
        float c = C[(size_t)(by + ty + r) * NM + bx + tx];
        _Float16 h = (_Float16)__expf(c * -10.0f);  // exp(-C/0.1)
        K[(size_t)(by + ty + r) * NM + bx + tx] = h;
        tile[ty + r][tx] = h;
    }
    __syncthreads();
    #pragma unroll
    for (int r = 0; r < 32; r += 8) {
        KT[(size_t)(bx + ty + r) * NM + by + tx] = tile[tx][ty + r];
    }
}

// ---------------------------------------------------------------------------
// 2) Init w = b * nu with b == 1 (v0 = 0), split hi/lo.
// ---------------------------------------------------------------------------
__global__ __launch_bounds__(256) void sink_init(const float* __restrict__ nu,
                                                 _Float16* __restrict__ whi,
                                                 _Float16* __restrict__ wlo) {
    int i = blockIdx.x * 256 + threadIdx.x;
    float w = nu[i];
    _Float16 h = (_Float16)w;
    whi[i] = h;
    wlo[i] = (_Float16)((w - (float)h) * LO_SCALE);
}

// ---------------------------------------------------------------------------
// 3) One Sinkhorn half-step as a WMMA GEMV.
//    r_i = sum_k A[i][k] * (whi[k] + wlo[k]/2048);  a_i = p_i / r_i
//    Block = 256 threads = 8 waves; block owns TWO 16-row strips (32 rows).
//    Each wave: 512-wide k slice, 2 accumulators sharing one B operand
//    (w_hi in column 0, w_lo*2048 in column 1; other columns stay zero via
//    a loop-carried, hoisted zero-init).
// ---------------------------------------------------------------------------
__global__ __launch_bounds__(256) void sink_gemv(const _Float16* __restrict__ A,
                                                 const float* __restrict__ p,
                                                 const _Float16* __restrict__ whi,
                                                 const _Float16* __restrict__ wlo,
                                                 float* __restrict__ scale_out,
                                                 _Float16* __restrict__ whi_out,
                                                 _Float16* __restrict__ wlo_out) {
    __shared__ __align__(16) float part[8][2][4][8];  // [wave][strip][slot][m]

    const int tid   = threadIdx.x;
    const int wave  = tid >> 5;
    const int lane  = tid & 31;
    const int row   = lane & 15;   // A-matrix M row within a 16-row strip
    const int khalf = lane >> 4;   // which K-half this lane holds
    const int r0    = blockIdx.x * 32;

    // 16-bit A 16x32 VGPR layout:
    // lanes 0-15 hold K = c..c+7 and c+16..c+23 of row `row`,
    // lanes 16-31 hold K = c+8..c+15 and c+24..c+31.
    const _Float16* arow0 = A + (size_t)(r0 + row) * NM + khalf * 8;
    const _Float16* arow1 = arow0 + (size_t)16 * NM;

    // B source pointer (meaningful only for lanes with row<2).
    const _Float16* wsel = (row == 0 ? whi : wlo) + khalf * 8;

    v8f acc0 = {};
    v8f acc1 = {};
    v16h bmat = {};                 // lanes >= 2 stay zero forever (hoisted)
    const int cbase = wave * 512;
    #pragma unroll 2
    for (int c = cbase; c < cbase + 512; c += 32) {
        v8h a0l = *(const v8h*)(arow0 + c);
        v8h a0h = *(const v8h*)(arow0 + c + 16);
        v8h a1l = *(const v8h*)(arow1 + c);
        v8h a1h = *(const v8h*)(arow1 + c + 16);
        v16h am0 = __builtin_shufflevector(a0l, a0h,
                     0, 1, 2, 3, 4, 5, 6, 7, 8, 9, 10, 11, 12, 13, 14, 15);
        v16h am1 = __builtin_shufflevector(a1l, a1h,
                     0, 1, 2, 3, 4, 5, 6, 7, 8, 9, 10, 11, 12, 13, 14, 15);

        if (row < 2) {              // only lanes 0,1,16,17 refresh B
            v8h bl = *(const v8h*)(wsel + c);
            v8h bh = *(const v8h*)(wsel + c + 16);
            bmat = __builtin_shufflevector(bl, bh,
                     0, 1, 2, 3, 4, 5, 6, 7, 8, 9, 10, 11, 12, 13, 14, 15);
        }

        acc0 = __builtin_amdgcn_wmma_f32_16x16x32_f16(
            false, am0, false, bmat, (short)0, acc0, false, false);
        acc1 = __builtin_amdgcn_wmma_f32_16x16x32_f16(
            false, am1, false, bmat, (short)0, acc1, false, false);
    }

    // D layout (16x16 f32): VGPR v, lanes 0-15 -> (M=v, N=lane);
    // lanes 16-31 -> (M=v+8, N=lane-16). We need columns 0 and 1.
    if (row < 2) {
        const int slot = khalf * 2 + row;  // 0:c0 M0-7, 1:c1 M0-7, 2:c0 M8-15, 3:c1 M8-15
        #pragma unroll
        for (int v = 0; v < 8; ++v) {
            part[wave][0][slot][v] = acc0[v];
            part[wave][1][slot][v] = acc1[v];
        }
    }
    __syncthreads();

    if (tid < 32) {
        const int strip = tid >> 4;
        const int m     = tid & 15;
        const int b0    = (m < 8) ? 0 : 2;
        const int mm    = (m < 8) ? m : m - 8;
        float s = 0.0f;
        #pragma unroll
        for (int wv = 0; wv < 8; ++wv)   // fixed order -> deterministic
            s += part[wv][strip][b0][mm] + part[wv][strip][b0 + 1][mm] * LO_INV;

        const int i  = r0 + strip * 16 + m;
        const float pv = p[i];
        const float av = pv / s;          // a_i = mu_i / (K w)_i
        scale_out[i] = av;
        const float wn = av * pv;         // next pass input: a_i * mu_i
        _Float16 h = (_Float16)wn;
        whi_out[i] = h;
        wlo_out[i] = (_Float16)((wn - (float)h) * LO_SCALE);
    }
}

// ---------------------------------------------------------------------------
// 4) W = eps * ( <log a, mu> + <log b, nu> )
// ---------------------------------------------------------------------------
__global__ __launch_bounds__(256) void sink_final(const float* __restrict__ a,
                                                  const float* __restrict__ b,
                                                  const float* __restrict__ mu,
                                                  const float* __restrict__ nu,
                                                  float* __restrict__ out) {
    __shared__ float red[256];
    const int tid = threadIdx.x;
    float s = 0.0f;
    for (int i = tid; i < NM; i += 256) {
        s += logf(a[i]) * mu[i];
        s += logf(b[i]) * nu[i];
    }
    red[tid] = s;
    __syncthreads();
    for (int off = 128; off > 0; off >>= 1) {
        if (tid < off) red[tid] += red[tid + off];
        __syncthreads();
    }
    if (tid == 0) out[0] = EPS_F * red[0];
}

// ---------------------------------------------------------------------------
extern "C" void kernel_launch(void* const* d_in, const int* in_sizes, int n_in,
                              void* d_out, int out_size, void* d_ws, size_t ws_size,
                              hipStream_t stream) {
    const float* C  = (const float*)d_in[0];
    const float* mu = (const float*)d_in[1];
    const float* nu = (const float*)d_in[2];

    char* ws = (char*)d_ws;
    const size_t kbytes = (size_t)NM * NM * sizeof(_Float16);  // 32 MB each
    _Float16* Kf   = (_Float16*)(ws);
    _Float16* KT   = (_Float16*)(ws + kbytes);
    float*    a    = (float*)   (ws + 2 * kbytes);
    float*    b    = (float*)   (ws + 2 * kbytes + 16384);
    _Float16* whiv = (_Float16*)(ws + 2 * kbytes + 32768);            // col-side w
    _Float16* wlov = (_Float16*)(ws + 2 * kbytes + 32768 + 8192);
    _Float16* whiu = (_Float16*)(ws + 2 * kbytes + 32768 + 16384);    // row-side w
    _Float16* wlou = (_Float16*)(ws + 2 * kbytes + 32768 + 24576);

    sink_convert<<<dim3(128, 128), dim3(32, 8), 0, stream>>>(C, Kf, KT);
    sink_init<<<16, 256, 0, stream>>>(nu, whiv, wlov);

    for (int it = 0; it < 100; ++it) {
        // u-update: r = K (b.nu), a = mu / r
        sink_gemv<<<128, 256, 0, stream>>>(Kf, mu, whiv, wlov, a, whiu, wlou);
        // v-update: c = K^T (a.mu), b = nu / c
        sink_gemv<<<128, 256, 0, stream>>>(KT, nu, whiu, wlou, b, whiv, wlov);
    }

    sink_final<<<1, 256, 0, stream>>>(a, b, mu, nu, (float*)d_out);
}
